// EmotionMamba_86638080295113
// MI455X (gfx1250) — matrix operand order repaired and checked by reading, not compiled, not run on hardware
//
#include <hip/hip_runtime.h>

// ---------------------------------------------------------------------------
// EmotionMamba on MI455X (gfx1250, wave32, WMMA).
// Memory-bound (~35 GFLOP over <1GB activation traffic @ 23.3 TB/s):
//  - GEMMs via v_wmma_f32_16x16x32_f16, weights pre-packed into per-lane
//    B-fragment layout (single 32B v16h load per fragment, L2-resident).
//  - Per-layer LN fused into the in-proj GEMM's LDS A-staging.
//  - Sequential S=30 Mamba scan spread over B*DI = 524288 independent lanes.
// ---------------------------------------------------------------------------

typedef __attribute__((ext_vector_type(16))) _Float16 v16h;
typedef __attribute__((ext_vector_type(8)))  _Float16 v8h;
typedef __attribute__((ext_vector_type(4)))  _Float16 v4h;
typedef __attribute__((ext_vector_type(8)))  float    v8f;

#define B_   2048
#define S_   30
#define DIN_ 512
#define H_   128
#define DI_  256
#define N_   16
#define R_   16
#define L_   2
#define K_   4
#define OUT_ 512
#define E_   7
#define ROWS_ (B_ * S_)   // 61440 tokens

// ---------------- workspace layout (float offsets) -------------------------
static const long long OFF_EMO = 0;                               // ROWS_*H_
static const long long OFF_XZ  = OFF_EMO + (long long)ROWS_*H_;   // ROWS_*2*DI_
static const long long OFF_XC  = OFF_XZ  + (long long)ROWS_*2*DI_;
static const long long OFF_DBC = OFF_XC  + (long long)ROWS_*DI_;  // ROWS_*48
static const long long OFF_DT  = OFF_DBC + (long long)ROWS_*48;
static const long long OFF_Y   = OFF_DT  + (long long)ROWS_*DI_;
static const long long OFF_PO  = OFF_Y   + (long long)ROWS_*DI_;  // B_*H_
static const long long OFF_H1  = OFF_PO  + (long long)B_*H_;      // B_*OUT_
static const long long OFF_F16 = OFF_H1  + (long long)B_*OUT_;    // packed f16 W

// packed f16 weight region (half offsets, based at OFF_F16); all 32B aligned
static const long long WH_PROJ = 0;                                // 512*128
static const long long WH_IN   = WH_PROJ + 512*128;                // L*128*512
static const long long WH_XP   = WH_IN   + (long long)L_*128*512;  // L*256*48
static const long long WH_OUT  = WH_XP   + (long long)L_*256*48;   // L*256*128
static const long long WH_FC1  = WH_OUT  + (long long)L_*256*128;  // 128*512

// ---------------------------------------------------------------------------
// Pack W[K,N] (f32, row-major) into WMMA B-fragment layout (f16):
//   out[((nt*(K/32)+kc)*32 + lane)*16 + e] = W[kc*32 + (lane>>4)*8 + (e&7) + (e>>3)*16][nt*16 + (lane&15)]
// One thread per packed half; writes fully coalesced.
// ---------------------------------------------------------------------------
__global__ __launch_bounds__(256) void pack_b_kernel(
    const float* __restrict__ W, _Float16* __restrict__ out, int K, int N) {
    const int idx  = blockIdx.x * 256 + threadIdx.x;    // total K*N
    const int e    = idx & 15;
    const int lane = (idx >> 4) & 31;
    const int rest = idx >> 9;
    const int kTiles = K >> 5;
    const int kc = rest % kTiles;
    const int nt = rest / kTiles;
    const int k = kc * 32 + (lane >> 4) * 8 + (e & 7) + (e >> 3) * 16;
    const int n = nt * 16 + (lane & 15);
    out[idx] = (_Float16)W[k * N + n];
}

// ---------------------------------------------------------------------------
// Fragment helpers
// ---------------------------------------------------------------------------
__device__ __forceinline__ v16h load_a_frag(const _Float16* ap) {
    // A fragment (ISA 7.12.2): elems 0..7 = K[base .. +7], 8..15 = K[base+16 .. +23]
    v8h lo = *(const v8h*)ap;
    v8h hi = *(const v8h*)(ap + 16);
    v16h af;
#pragma unroll
    for (int i = 0; i < 8; ++i) { af[i] = lo[i]; af[8 + i] = hi[i]; }
    return af;
}

// ---------------------------------------------------------------------------
// Generic row-tile WMMA GEMM: C[M,N] = A[M,K] * W[K,N] (+bias) (+residual)
// Workgroup = 8 waves, one 16-row M tile; wave w handles n-tiles w, w+8, ...
// W is pre-packed f16 fragments; one aligned v16h load per WMMA.
// EPI==0: C = A*W + bias ; EPI==1: C += A*W + bias
// ---------------------------------------------------------------------------
template <int EPI>
__global__ __launch_bounds__(256) void gemm_wmma_kernel(
    const float* __restrict__ A, int lda,
    const _Float16* __restrict__ Wp,
    const float* __restrict__ bias,
    float* __restrict__ C, int K, int N) {
    __shared__ _Float16 ldsA[16 * 512];   // up to K=512

    const int tid  = threadIdx.x;
    const int lane = tid & 31;
    const int wave = tid >> 5;
    const int rowBase = blockIdx.x * 16;

    // A tile stage: float4 global loads, v4h LDS stores (all 16B/8B aligned)
    const int total = 16 * K;
    for (int idx = tid * 4; idx < total; idx += 1024) {
        const int r = idx / K, c = idx - r * K;
        const float4 v = *(const float4*)&A[(rowBase + r) * lda + c];
        v4h h4; h4[0] = (_Float16)v.x; h4[1] = (_Float16)v.y;
                h4[2] = (_Float16)v.z; h4[3] = (_Float16)v.w;
        *(v4h*)&ldsA[idx] = h4;
    }
    __syncthreads();

    const int half = lane >> 4;
    const int l15  = lane & 15;
    const int nTiles = N >> 4;
    const int kTiles = K >> 5;
    const v16h* W16 = (const v16h*)Wp;

    for (int nt = wave; nt < nTiles; nt += 8) {
        v8f acc = {};
        const int n = nt * 16 + l15;
        const v16h* wrow = W16 + nt * kTiles * 32 + lane;
        for (int kc = 0; kc < kTiles; ++kc) {
            const v16h af = load_a_frag(&ldsA[l15 * K + kc * 32 + half * 8]);
            const v16h bf = wrow[kc * 32];
            acc = __builtin_amdgcn_wmma_f32_16x16x32_f16(false, af, false, bf,
                                                         (short)0, acc, false, false);
        }
        const float bv = bias ? bias[n] : 0.0f;
#pragma unroll
        for (int v = 0; v < 8; ++v) {
            const int m = half * 8 + v;           // D row = 8*half + v
            const int o = (rowBase + m) * N + n;
            float r = acc[v] + bv;
            if (EPI == 1) r += C[o];
            C[o] = r;
        }
    }
}

// ---------------------------------------------------------------------------
// Fused LN(A) -> GEMM: C = LN(A) @ W + bias ; K=128 fixed (Mamba in-proj).
// LN is applied on the staged 16x128 tile in LDS before fragment extraction.
// ---------------------------------------------------------------------------
__global__ __launch_bounds__(256) void gemm_ln_wmma_kernel(
    const float* __restrict__ A,
    const _Float16* __restrict__ Wp,
    const float* __restrict__ bias,
    const float* __restrict__ g, const float* __restrict__ bvec,
    float* __restrict__ C, int N) {
    __shared__ float    ldsF[16 * 128];
    __shared__ _Float16 ldsA[16 * 128];
    __shared__ float    mu[16], rs[16];

    const int tid  = threadIdx.x;
    const int lane = tid & 31;
    const int wave = tid >> 5;
    const int rowBase = blockIdx.x * 16;

    for (int idx = tid * 4; idx < 16 * 128; idx += 1024) {
        *(float4*)&ldsF[idx] = *(const float4*)&A[rowBase * 128 + idx];
    }
    __syncthreads();

    if (tid < 16) {
        float s = 0.f, s2 = 0.f;
        const float* p = &ldsF[tid * 128];
        for (int i = 0; i < 128; ++i) { float v = p[i]; s += v; s2 += v * v; }
        const float m = s * (1.0f / 128.0f);
        mu[tid] = m;
        rs[tid] = rsqrtf(s2 * (1.0f / 128.0f) - m * m + 1e-5f);
    }
    __syncthreads();

    for (int idx = tid; idx < 16 * 128; idx += 256) {
        const int r = idx >> 7, c = idx & 127;
        ldsA[idx] = (_Float16)((ldsF[idx] - mu[r]) * rs[r] * g[c] + bvec[c]);
    }
    __syncthreads();

    const int half = lane >> 4;
    const int l15  = lane & 15;
    const int nTiles = N >> 4;
    const v16h* W16 = (const v16h*)Wp;

    for (int nt = wave; nt < nTiles; nt += 8) {
        v8f acc = {};
        const int n = nt * 16 + l15;
        const v16h* wrow = W16 + nt * 4 * 32 + lane;   // kTiles = 128/32 = 4
#pragma unroll
        for (int kc = 0; kc < 4; ++kc) {
            const v16h af = load_a_frag(&ldsA[l15 * 128 + kc * 32 + half * 8]);
            const v16h bf = wrow[kc * 32];
            acc = __builtin_amdgcn_wmma_f32_16x16x32_f16(false, af, false, bf,
                                                         (short)0, acc, false, false);
        }
        const float bv = bias[n];
#pragma unroll
        for (int v = 0; v < 8; ++v) {
            C[(rowBase + half * 8 + v) * N + n] = acc[v] + bv;
        }
    }
}

// ---------------------------------------------------------------------------
// Fused input projection + LN0: emo = LN(X @ Wproj + b) ; K=512, N=128
// ---------------------------------------------------------------------------
__global__ __launch_bounds__(256) void proj_ln_kernel(
    const float* __restrict__ X, const _Float16* __restrict__ Wp,
    const float* __restrict__ bias, const float* __restrict__ g,
    const float* __restrict__ bv, float* __restrict__ out) {
    __shared__ _Float16 ldsA[16 * 512];
    __shared__ float    ldsC[16 * 128];
    __shared__ float    mu[16], rs[16];

    const int tid  = threadIdx.x;
    const int lane = tid & 31;
    const int wave = tid >> 5;
    const int rowBase = blockIdx.x * 16;

    for (int idx = tid * 4; idx < 16 * 512; idx += 1024) {
        const float4 v = *(const float4*)&X[rowBase * 512 + idx];
        v4h h4; h4[0] = (_Float16)v.x; h4[1] = (_Float16)v.y;
                h4[2] = (_Float16)v.z; h4[3] = (_Float16)v.w;
        *(v4h*)&ldsA[idx] = h4;
    }
    __syncthreads();

    const int half = lane >> 4;
    const int l15  = lane & 15;
    const v16h* W16 = (const v16h*)Wp;
    {
        v8f acc = {};
        const int n = wave * 16 + l15;            // 8 waves cover N=128 exactly
        const v16h* wrow = W16 + wave * 16 * 32 + lane;  // kTiles = 512/32 = 16
#pragma unroll 4
        for (int kc = 0; kc < 16; ++kc) {
            const v16h af = load_a_frag(&ldsA[l15 * 512 + kc * 32 + half * 8]);
            const v16h bf = wrow[kc * 32];
            acc = __builtin_amdgcn_wmma_f32_16x16x32_f16(false, af, false, bf,
                                                         (short)0, acc, false, false);
        }
        const float bb = bias[n];
#pragma unroll
        for (int v = 0; v < 8; ++v) ldsC[(half * 8 + v) * 128 + n] = acc[v] + bb;
    }
    __syncthreads();

    if (tid < 16) {
        float s = 0.f, s2 = 0.f;
        const float* p = &ldsC[tid * 128];
        for (int i = 0; i < 128; ++i) { float v = p[i]; s += v; s2 += v * v; }
        const float m = s * (1.0f / 128.0f);
        mu[tid] = m;
        rs[tid] = rsqrtf(s2 * (1.0f / 128.0f) - m * m + 1e-5f);
    }
    __syncthreads();

    for (int idx = tid; idx < 16 * 128; idx += 256) {
        const int r = idx >> 7, c = idx & 127;
        out[(rowBase + r) * 128 + c] = (ldsC[idx] - mu[r]) * rs[r] * g[c] + bv[c];
    }
}

// ---------------------------------------------------------------------------
// Depthwise causal conv (K=4) over S + SiLU.  xin = xz[..., :DI]
// ---------------------------------------------------------------------------
__global__ __launch_bounds__(256) void conv_silu_kernel(
    const float* __restrict__ xz, const float* __restrict__ cw,
    const float* __restrict__ cb, float* __restrict__ xc) {
    const int idx = blockIdx.x * 256 + threadIdx.x;   // ROWS_*DI_ threads
    const int d  = idx & (DI_ - 1);
    const int bs = idx >> 8;
    const int s  = bs % S_;
    const int b  = bs / S_;
    float acc = cb[d];
#pragma unroll
    for (int k = 0; k < K_; ++k) {
        const int sp = s + k - (K_ - 1);
        if (sp >= 0) acc += xz[(b * S_ + sp) * (2 * DI_) + d] * cw[d * K_ + k];
    }
    xc[idx] = acc / (1.0f + __expf(-acc));            // SiLU
}

// ---------------------------------------------------------------------------
// dt = softplus(dbc[:, :R] @ dt_w + dt_b)
// ---------------------------------------------------------------------------
__global__ __launch_bounds__(256) void dt_kernel(
    const float* __restrict__ dbc, const float* __restrict__ dt_w,
    const float* __restrict__ dt_b, float* __restrict__ dt) {
    const int idx = blockIdx.x * 256 + threadIdx.x;
    const int d  = idx & (DI_ - 1);
    const int bs = idx >> 8;
    const float* p = dbc + bs * 48;
    float a = dt_b[d];
#pragma unroll
    for (int r = 0; r < R_; ++r) a += p[r] * dt_w[r * DI_ + d];
    dt[idx] = (a > 20.0f) ? a : log1pf(__expf(a));
}

// ---------------------------------------------------------------------------
// Selective scan over S=30 (sequential), one lane per (b,d), N=16 states in
// registers; fused +x*D and z-gate.  y = (scan + xc*D) * silu(z)
// ---------------------------------------------------------------------------
__global__ __launch_bounds__(256) void scan_kernel(
    const float* __restrict__ dt, const float* __restrict__ xc,
    const float* __restrict__ dbc, const float* __restrict__ xz,
    const float* __restrict__ A_log, const float* __restrict__ Dp,
    float* __restrict__ y) {
    const int t = blockIdx.x * 256 + threadIdx.x;     // B_*DI_ threads
    const int d = t & (DI_ - 1);
    const int b = t >> 8;
    float a[N_], h[N_];
#pragma unroll
    for (int n = 0; n < N_; ++n) { a[n] = -__expf(A_log[d * N_ + n]); h[n] = 0.f; }
    const float Dd = Dp[d];
    for (int s = 0; s < S_; ++s) {
        const int bs = b * S_ + s;
        const float dtv = dt[bs * DI_ + d];
        const float xv  = xc[bs * DI_ + d];
        const float dx  = dtv * xv;
        const float* bc = dbc + bs * 48;
        float yv = 0.f;
#pragma unroll
        for (int n = 0; n < N_; ++n) {
            h[n] = __expf(dtv * a[n]) * h[n] + dx * bc[16 + n];
            yv  += h[n] * bc[32 + n];
        }
        const float zv = xz[bs * (2 * DI_) + DI_ + d];
        y[bs * DI_ + d] = (yv + xv * Dd) * (zv / (1.0f + __expf(-zv)));
    }
}

// ---------------------------------------------------------------------------
// Mean-pool over S
// ---------------------------------------------------------------------------
__global__ __launch_bounds__(256) void pool_kernel(
    const float* __restrict__ emo, float* __restrict__ pooled) {
    const int t = blockIdx.x * 256 + threadIdx.x;     // B_*H_ threads
    const int hh = t & (H_ - 1);
    const int b  = t >> 7;
    float s = 0.f;
    for (int i = 0; i < S_; ++i) s += emo[(b * S_ + i) * H_ + hh];
    pooled[t] = s * (1.0f / (float)S_);
}

// ---------------------------------------------------------------------------
// Final: LN(h1) -> gelu(tanh) -> fc2 ; one block per batch element
// ---------------------------------------------------------------------------
__global__ __launch_bounds__(256) void final_kernel(
    const float* __restrict__ h1, const float* __restrict__ g,
    const float* __restrict__ bv, const float* __restrict__ w2,
    const float* __restrict__ b2, float* __restrict__ out) {
    __shared__ float buf[OUT_];
    __shared__ float red[256], red2[256];
    __shared__ float sMean, sRstd;
    const int tid = threadIdx.x;
    const int b   = blockIdx.x;
    float s = 0.f, s2 = 0.f;
#pragma unroll
    for (int i = 0; i < 2; ++i) {
        const int c = tid + 256 * i;
        float v = h1[b * OUT_ + c];
        buf[c] = v; s += v; s2 += v * v;
    }
    red[tid] = s; red2[tid] = s2;
    __syncthreads();
    for (int off = 128; off > 0; off >>= 1) {
        if (tid < off) { red[tid] += red[tid + off]; red2[tid] += red2[tid + off]; }
        __syncthreads();
    }
    if (tid == 0) {
        float m = red[0] * (1.0f / OUT_);
        sMean = m;
        sRstd = rsqrtf(red2[0] * (1.0f / OUT_) - m * m + 1e-5f);
    }
    __syncthreads();
#pragma unroll
    for (int i = 0; i < 2; ++i) {
        const int c = tid + 256 * i;
        float x = (buf[c] - sMean) * sRstd * g[c] + bv[c];
        // tanh-approx GELU (jax.nn.gelu default)
        buf[c] = 0.5f * x * (1.0f + tanhf(0.7978845608f * (x + 0.044715f * x * x * x)));
    }
    __syncthreads();
    if (tid < E_) {
        float acc = b2[tid];
        for (int o = 0; o < OUT_; ++o) acc += buf[o] * w2[o * E_ + tid];
        out[b * E_ + tid] = acc;
    }
}

// ---------------------------------------------------------------------------
extern "C" void kernel_launch(void* const* d_in, const int* in_sizes, int n_in,
                              void* d_out, int out_size, void* d_ws, size_t ws_size,
                              hipStream_t stream) {
    (void)in_sizes; (void)n_in; (void)out_size; (void)ws_size;
    const float* X      = (const float*)d_in[0];
    const float* proj_w = (const float*)d_in[1];
    const float* proj_b = (const float*)d_in[2];
    const float* ln0_g  = (const float*)d_in[3];
    const float* ln0_b  = (const float*)d_in[4];
    const float* mlng   = (const float*)d_in[5];
    const float* mlnb   = (const float*)d_in[6];
    const float* in_w   = (const float*)d_in[7];
    const float* in_b   = (const float*)d_in[8];
    const float* conv_w = (const float*)d_in[9];
    const float* conv_b = (const float*)d_in[10];
    const float* xp_w   = (const float*)d_in[11];
    const float* dt_w   = (const float*)d_in[12];
    const float* dt_b   = (const float*)d_in[13];
    const float* A_log  = (const float*)d_in[14];
    const float* Dp     = (const float*)d_in[15];
    const float* out_w  = (const float*)d_in[16];
    const float* out_b  = (const float*)d_in[17];
    const float* fc1_w  = (const float*)d_in[18];
    const float* fc1_b  = (const float*)d_in[19];
    const float* lnf_g  = (const float*)d_in[20];
    const float* lnf_b  = (const float*)d_in[21];
    const float* fc2_w  = (const float*)d_in[22];
    const float* fc2_b  = (const float*)d_in[23];

    float* ws     = (float*)d_ws;
    float* emo    = ws + OFF_EMO;
    float* xz     = ws + OFF_XZ;
    float* xc     = ws + OFF_XC;
    float* dbc    = ws + OFF_DBC;
    float* dtbuf  = ws + OFF_DT;
    float* ybuf   = ws + OFF_Y;
    float* pooled = ws + OFF_PO;
    float* h1     = ws + OFF_H1;
    _Float16* wh  = (_Float16*)(ws + OFF_F16);

    _Float16* proj_wh = wh + WH_PROJ;
    _Float16* in_wh   = wh + WH_IN;
    _Float16* xp_wh   = wh + WH_XP;
    _Float16* out_wh  = wh + WH_OUT;
    _Float16* fc1_wh  = wh + WH_FC1;

    // ---- pack weights into WMMA B-fragment layout (f32 -> f16) ----
    pack_b_kernel<<<(512 * 128) / 256, 256, 0, stream>>>(proj_w, proj_wh, 512, 128);
    for (int l = 0; l < L_; ++l) {
        pack_b_kernel<<<(128 * 512) / 256, 256, 0, stream>>>(
            in_w + l * 128 * 512, in_wh + l * 128 * 512, 128, 512);
        pack_b_kernel<<<(256 * 48) / 256, 256, 0, stream>>>(
            xp_w + l * 256 * 48, xp_wh + l * 256 * 48, 256, 48);
        pack_b_kernel<<<(256 * 128) / 256, 256, 0, stream>>>(
            out_w + l * 256 * 128, out_wh + l * 256 * 128, 256, 128);
    }
    pack_b_kernel<<<(128 * 512) / 256, 256, 0, stream>>>(fc1_w, fc1_wh, 128, 512);

    const int mTiles = ROWS_ / 16;                 // 3840
    // ---- input projection + LN0 ----
    proj_ln_kernel<<<mTiles, 256, 0, stream>>>(X, proj_wh, proj_b, ln0_g, ln0_b, emo);

    // ---- Mamba layers ----
    for (int l = 0; l < L_; ++l) {
        gemm_ln_wmma_kernel<<<mTiles, 256, 0, stream>>>(
            emo, in_wh + l * 128 * 512, in_b + l * 2 * DI_,
            mlng + l * H_, mlnb + l * H_, xz, 2 * DI_);
        conv_silu_kernel<<<ROWS_ * DI_ / 256, 256, 0, stream>>>(
            xz, conv_w + l * DI_ * K_, conv_b + l * DI_, xc);
        gemm_wmma_kernel<0><<<mTiles, 256, 0, stream>>>(
            xc, DI_, xp_wh + l * 256 * 48, nullptr, dbc, DI_, 48);
        dt_kernel<<<ROWS_ * DI_ / 256, 256, 0, stream>>>(
            dbc, dt_w + l * R_ * DI_, dt_b + l * DI_, dtbuf);
        scan_kernel<<<B_ * DI_ / 256, 256, 0, stream>>>(
            dtbuf, xc, dbc, xz, A_log + l * DI_ * N_, Dp + l * DI_, ybuf);
        gemm_wmma_kernel<1><<<mTiles, 256, 0, stream>>>(
            ybuf, DI_, out_wh + l * 256 * 128, out_b + l * H_, emo, DI_, H_);
    }

    // ---- head ----
    pool_kernel<<<B_ * H_ / 256, 256, 0, stream>>>(emo, pooled);
    gemm_wmma_kernel<0><<<B_ / 16, 256, 0, stream>>>(pooled, H_, fc1_wh, fc1_b, h1, H_, OUT_);
    final_kernel<<<B_, 256, 0, stream>>>(h1, lnf_g, lnf_b, fc2_w, fc2_b, (float*)d_out);
}
